// WATModel_30623116820688
// MI455X (gfx1250) — matrix operand order repaired and checked by reading, not compiled, not run on hardware
//
#include <hip/hip_runtime.h>
#include <hip/hip_bf16.h>
#include <math.h>

// ---------------------------------------------------------------------------
// Model dims (fixed by the reference)
// ---------------------------------------------------------------------------
#define S_LEN 2048
#define D_DIM 768
#define V_DIM 32000
#define K_TAPS 3
#define EPSF 1.1920929e-07f

typedef __attribute__((ext_vector_type(16))) __bf16 v16bf;
typedef __attribute__((ext_vector_type(8)))  float  v8f;
typedef __attribute__((ext_vector_type(4)))  float  f32x4;

union Frag32B { v16bf v; f32x4 q[2]; };   // 32 bytes: 16 bf16 lanes-worth

__device__ __forceinline__ unsigned short f2bf(float f) {
    // round-to-nearest-even fp32 -> bf16
    unsigned int u = __float_as_uint(f);
    unsigned int r = (u + 0x7FFFu + ((u >> 16) & 1u)) >> 16;
    return (unsigned short)r;
}
__device__ __forceinline__ float sigm(float x) { return 1.0f / (1.0f + __expf(-x)); }

// ---------------------------------------------------------------------------
// Weight packing: fp32 [K,N]-style weight -> WMMA-native packed-B (bf16).
// Tile (nt, kt) of 32(K)x16(N) stored as 512 contiguous bf16, lane-major:
// dst[((nt*nKT + kt)*32 + lane)*16 + i] = W[kt*32 + (lane>>4)*8 + (i&7) + (i>>3)*16][nt*16 + (lane&15)]
// (ISA 16-bit B-matrix 32x16 layout; each lane loads one contiguous 32B chunk)
// Generic source addressing: kk -> (ko = kk/kSplit, ki = kk%kSplit);
// src index = ko*soK + ki*soI + nn*sn.  Handles plain weights and conv taps.
// ---------------------------------------------------------------------------
__global__ void pack_weight(const float* __restrict__ src, unsigned short* __restrict__ dst,
                            int Kdim, int Ndim, int kSplit, long soK, long soI, long sn)
{
    long idx = (long)blockIdx.x * blockDim.x + threadIdx.x;
    long total = (long)Kdim * Ndim;
    if (idx >= total) return;
    int  nKT    = Kdim >> 5;
    long tile   = idx >> 9;            // / 512
    int  within = (int)(idx & 511);
    int  lane   = within >> 4;
    int  i      = within & 15;
    long nt = tile / nKT;
    long kt = tile % nKT;
    long kk = kt * 32 + (long)(lane >> 4) * 8 + (i & 7) + (i >> 3) * 16;
    long nn = nt * 16 + (lane & 15);
    long ko = kk / kSplit;
    long ki = kk % kSplit;
    dst[idx] = f2bf(src[ko * soK + ki * soI + nn * sn]);
}

// ---------------------------------------------------------------------------
// h = embedding[x] + pos_emb  (directly to bf16)
// ---------------------------------------------------------------------------
__global__ void embed_kernel(const int* __restrict__ x, const float* __restrict__ emb,
                             const float* __restrict__ pos, unsigned short* __restrict__ hbf)
{
    long idx = (long)blockIdx.x * blockDim.x + threadIdx.x;
    if (idx >= (long)S_LEN * D_DIM) return;
    long t = idx / D_DIM, d = idx % D_DIM;
    long tok = (long)x[t];
    hbf[idx] = f2bf(emb[tok * D_DIM + d] + pos[idx]);
}

// im2col for the causal K=3 conv: X3[t][k*768+d] = h[t+k-2][d] (zero-pad left)
__global__ void im2col_kernel(const unsigned short* __restrict__ hbf, unsigned short* __restrict__ x3)
{
    long idx = (long)blockIdx.x * blockDim.x + threadIdx.x;
    if (idx >= (long)S_LEN * D_DIM * K_TAPS) return;
    long t = idx / (D_DIM * K_TAPS);
    int  j = (int)(idx % (D_DIM * K_TAPS));
    int  k = j / D_DIM, d = j % D_DIM;
    long tt = t + k - (K_TAPS - 1);
    x3[idx] = (tt >= 0) ? hbf[tt * D_DIM + d] : (unsigned short)0;
}

// ---------------------------------------------------------------------------
// Generic bf16 WMMA GEMM:  C[M,N] (f32) = A[M,K](bf16, row-major lda) * Bpacked + bias
// Wave tile: 32(M) x 64(N): 2 A frags x 4 B frags -> 8x v_wmma_f32_16x16x32_bf16
// per K-step (B reused 2x, A 4x).  Register double-buffering: iteration kt+1's
// A/B fragments are loaded before iteration kt's WMMAs issue, so VMEM latency
// hides under the matrix ops (partial s_wait_loadcnt instead of 0).
// Block: 128 threads = 4 waves = 64(M) x 128(N).  Requires N % 128 == 0, K % 32 == 0.
// A per-lane load: two aligned 16B chunks (ISA 16-bit A 16x32 layout).
// B per-lane load: one contiguous 32B chunk from the packed buffer.
// ---------------------------------------------------------------------------
__global__ __launch_bounds__(128) void gemm_bf16_wmma(
    const unsigned short* __restrict__ A, long lda,
    const unsigned short* __restrict__ Bp,
    const float* __restrict__ bias,
    float* __restrict__ C, long ldc,
    int M, int nKT)
{
    const int lane = threadIdx.x & 31;
    const int wave = threadIdx.x >> 5;
    const int hi   = lane >> 4;
    const int lm   = lane & 15;

    const long m0     = (long)blockIdx.y * 64 + (long)(wave & 1) * 32;
    const long ntBase = (long)blockIdx.x * 8  + (long)(wave >> 1) * 4;

    long row0 = m0 + lm;
    long row1 = m0 + 16 + lm;
    long rowLd0 = (row0 < M) ? row0 : (long)(M - 1);   // clamp; stores are guarded
    long rowLd1 = (row1 < M) ? row1 : (long)(M - 1);
    const unsigned short* aRow0 = A + rowLd0 * lda + hi * 8;
    const unsigned short* aRow1 = A + rowLd1 * lda + hi * 8;
    const unsigned short* bBase = Bp + ntBase * (long)nKT * 512 + (long)lane * 16;
    const long bStride = (long)nKT * 512;

    v8f acc[2][4] = {};

    // preload kt = 0
    Frag32B a0[2], b0[4];
    a0[0].q[0] = *(const f32x4*)(aRow0);
    a0[0].q[1] = *(const f32x4*)(aRow0 + 16);
    a0[1].q[0] = *(const f32x4*)(aRow1);
    a0[1].q[1] = *(const f32x4*)(aRow1 + 16);
    #pragma unroll
    for (int j = 0; j < 4; ++j) {
        const unsigned short* bp = bBase + (long)j * bStride;
        b0[j].q[0] = *(const f32x4*)(bp);
        b0[j].q[1] = *(const f32x4*)(bp + 8);
    }

    for (int kt = 0; kt < nKT; ++kt) {
        // issue next iteration's loads first (last iter: harmless reload of kt)
        const int ktn = (kt + 1 < nKT) ? (kt + 1) : kt;
        Frag32B a1[2], b1[4];
        a1[0].q[0] = *(const f32x4*)(aRow0 + (long)ktn * 32);
        a1[0].q[1] = *(const f32x4*)(aRow0 + (long)ktn * 32 + 16);
        a1[1].q[0] = *(const f32x4*)(aRow1 + (long)ktn * 32);
        a1[1].q[1] = *(const f32x4*)(aRow1 + (long)ktn * 32 + 16);
        #pragma unroll
        for (int j = 0; j < 4; ++j) {
            const unsigned short* bp = bBase + (long)j * bStride + (long)ktn * 512;
            b1[j].q[0] = *(const f32x4*)(bp);
            b1[j].q[1] = *(const f32x4*)(bp + 8);
        }
        if (kt + 2 < nKT)   // keep the B stream warm in near caches
            __builtin_prefetch(bBase + (long)(kt + 2) * 512, 0, 3);

        // 8 WMMAs on the current (already resident) fragments
        #pragma unroll
        for (int j = 0; j < 4; ++j) {
            acc[0][j] = __builtin_amdgcn_wmma_f32_16x16x32_bf16(
                            false, a0[0].v, false, b0[j].v, (short)0, acc[0][j], false, false);
            acc[1][j] = __builtin_amdgcn_wmma_f32_16x16x32_bf16(
                            false, a0[1].v, false, b0[j].v, (short)0, acc[1][j], false, false);
        }

        a0[0] = a1[0]; a0[1] = a1[1];
        #pragma unroll
        for (int j = 0; j < 4; ++j) b0[j] = b1[j];
    }

    #pragma unroll
    for (int j = 0; j < 4; ++j) {
        long n0 = (ntBase + j) * 16 + lm;
        float bv = bias ? bias[n0] : 0.0f;
        #pragma unroll
        for (int i = 0; i < 2; ++i) {
            #pragma unroll
            for (int r = 0; r < 8; ++r) {
                long rr = m0 + (long)i * 16 + (long)hi * 8 + r;  // C/D: VGPR r -> row r (+8 hi half)
                if (rr < M) C[rr * ldc + n0] = acc[i][j][r] + bv;
            }
        }
    }
}

// ---------------------------------------------------------------------------
// Elementwise kernels
// ---------------------------------------------------------------------------
__global__ void f32_to_bf16_kernel(const float* __restrict__ src, unsigned short* __restrict__ dst, long n)
{
    long idx = (long)blockIdx.x * blockDim.x + threadIdx.x;
    if (idx < n) dst[idx] = f2bf(src[idx]);
}

// nodes = conv_out * sigmoid(glin)
__global__ void nodes_kernel(const float* __restrict__ conv_out, const float* __restrict__ glin,
                             float* __restrict__ curr)
{
    long idx = (long)blockIdx.x * blockDim.x + threadIdx.x;
    if (idx >= (long)S_LEN * D_DIM) return;
    curr[idx] = conv_out[idx] * sigm(glin[idx]);
}

// comb[m] = [curr[m] | curr[m+step]]  (bf16, width 1536)
__global__ void comb_kernel(const float* __restrict__ curr, unsigned short* __restrict__ comb,
                            int Sm, int step)
{
    long idx = (long)blockIdx.x * blockDim.x + threadIdx.x;
    if (idx >= (long)Sm * (2 * D_DIM)) return;
    long m = idx / (2 * D_DIM);
    int  j = (int)(idx % (2 * D_DIM));
    float v = (j < D_DIM) ? curr[m * D_DIM + j]
                          : curr[(m + step) * D_DIM + (j - D_DIM)];
    comb[idx] = f2bf(v);
}

// per-row merge epilogue: sigmoid gates + RMSNorm + highway + residual
__global__ __launch_bounds__(256) void merge_kernel(
    const float* __restrict__ cur, float* __restrict__ nxt,
    const float* __restrict__ val, const float* __restrict__ gl,
    const float* __restrict__ rgl, const float* __restrict__ hgl,
    const float* __restrict__ rmsw, int step)
{
    const int m   = blockIdx.x;
    const int tid = threadIdx.x;
    __shared__ float red[256];

    float vloc[3];
    float ss = 0.0f;
    #pragma unroll
    for (int j = 0; j < 3; ++j) {
        int  c = tid + j * 256;
        long o = (long)m * D_DIM + c;
        float vv = val[o] * sigm(gl[o]);
        vloc[j] = vv;
        ss += vv * vv;
    }
    red[tid] = ss;
    __syncthreads();
    for (int s = 128; s > 0; s >>= 1) {
        if (tid < s) red[tid] += red[tid + s];
        __syncthreads();
    }
    const float rinv = rsqrtf(red[0] * (1.0f / (float)D_DIM) + EPSF);

    #pragma unroll
    for (int j = 0; j < 3; ++j) {
        int  c  = tid + j * 256;
        long o  = (long)m * D_DIM + c;
        long oR = (long)(m + step) * D_DIM + c;
        float merged = vloc[j] * rinv * rmsw[c];
        float r = cur[oR];
        float l = cur[o];
        float hg = sigm(hgl[o]);
        merged = hg * merged + (1.0f - hg) * r;
        float rg = sigm(rgl[o]);
        merged = rg * merged + (1.0f - rg) * 0.5f * (l + r);
        nxt[oR] = merged;
    }
}

// ---------------------------------------------------------------------------
// Host orchestration
// ---------------------------------------------------------------------------
extern "C" void kernel_launch(void* const* d_in, const int* in_sizes, int n_in,
                              void* d_out, int out_size, void* d_ws, size_t ws_size,
                              hipStream_t stream)
{
    (void)in_sizes; (void)n_in; (void)out_size; (void)ws_size;

    const int*   x     = (const int*)d_in[0];
    const float* emb   = (const float*)d_in[1];
    const float* pos   = (const float*)d_in[2];
    const float* convw = (const float*)d_in[3];
    const float* convb = (const float*)d_in[4];
    const float* wg1   = (const float*)d_in[5];
    const float* bg1   = (const float*)d_in[6];
    const float* wmv   = (const float*)d_in[7];
    const float* bmv   = (const float*)d_in[8];
    const float* wmg   = (const float*)d_in[9];
    const float* bmg   = (const float*)d_in[10];
    const float* wrg   = (const float*)d_in[11];
    const float* brg   = (const float*)d_in[12];
    const float* whw   = (const float*)d_in[13];
    const float* bhw   = (const float*)d_in[14];
    const float* rmsw  = (const float*)d_in[15];
    const float* wpred = (const float*)d_in[16];
    const float* bpred = (const float*)d_in[17];
    float*       out   = (float*)d_out;

    // ---- carve workspace (256B aligned) ----
    char*  ws  = (char*)d_ws;
    size_t off = 0;
    auto carve = [&](size_t bytes) -> char* {
        char* p = ws + off;
        off = (off + bytes + 255) & ~(size_t)255;
        return p;
    };
    unsigned short* wg1_p = (unsigned short*)carve((size_t)D_DIM * D_DIM * 2);
    unsigned short* wmv_p = (unsigned short*)carve((size_t)2 * D_DIM * D_DIM * 2);
    unsigned short* wmg_p = (unsigned short*)carve((size_t)2 * D_DIM * D_DIM * 2);
    unsigned short* wrg_p = (unsigned short*)carve((size_t)2 * D_DIM * D_DIM * 2);
    unsigned short* whw_p = (unsigned short*)carve((size_t)D_DIM * D_DIM * 2);
    unsigned short* wcv_p = (unsigned short*)carve((size_t)K_TAPS * D_DIM * D_DIM * 2);
    unsigned short* wpr_p = (unsigned short*)carve((size_t)D_DIM * V_DIM * 2);
    unsigned short* hbf   = (unsigned short*)carve((size_t)S_LEN * D_DIM * 2);            // reused as act-bf16
    unsigned short* x3    = (unsigned short*)carve((size_t)S_LEN * D_DIM * K_TAPS * 2);   // reused as comb
    float* conv_out = (float*)carve((size_t)S_LEN * D_DIM * 4);
    float* gl_buf   = (float*)carve((size_t)S_LEN * D_DIM * 4);                           // glin, then gate-lin
    float* val_buf  = (float*)carve((size_t)S_LEN * D_DIM * 4);
    float* rgl_buf  = (float*)carve((size_t)S_LEN * D_DIM * 4);
    float* hgl_buf  = (float*)carve((size_t)S_LEN * D_DIM * 4);
    float* curA     = (float*)carve((size_t)S_LEN * D_DIM * 4);
    float* curB     = (float*)carve((size_t)S_LEN * D_DIM * 4);

    auto pack = [&](const float* src, unsigned short* dst, int Kd, int Nd,
                    int kSplit, long soK, long soI, long sn) {
        long total = (long)Kd * Nd;
        pack_weight<<<dim3((unsigned)((total + 255) / 256)), dim3(256), 0, stream>>>(
            src, dst, Kd, Nd, kSplit, soK, soI, sn);
    };
    auto gemm = [&](const unsigned short* A, long lda, const unsigned short* Bp,
                    const float* bias, float* C, long ldc, int M, int N, int Kd) {
        dim3 grid((unsigned)(N / 128), (unsigned)((M + 63) / 64));
        gemm_bf16_wmma<<<grid, dim3(128), 0, stream>>>(A, lda, Bp, bias, C, ldc, M, Kd / 32);
    };

    // ---- pack all weights to WMMA layout (bf16) ----
    pack(wg1,   wg1_p, D_DIM,     D_DIM, D_DIM,     0, D_DIM, 1);
    pack(wmv,   wmv_p, 2 * D_DIM, D_DIM, 2 * D_DIM, 0, D_DIM, 1);
    pack(wmg,   wmg_p, 2 * D_DIM, D_DIM, 2 * D_DIM, 0, D_DIM, 1);
    pack(wrg,   wrg_p, 2 * D_DIM, D_DIM, 2 * D_DIM, 0, D_DIM, 1);
    pack(whw,   whw_p, D_DIM,     D_DIM, D_DIM,     0, D_DIM, 1);
    // Wcat[k*768 + i][o] = conv_w[o, i, k]  (conv_w flat: o*2304 + i*3 + k)
    pack(convw, wcv_p, K_TAPS * D_DIM, D_DIM, D_DIM, 1, K_TAPS, (long)K_TAPS * D_DIM);
    pack(wpred, wpr_p, D_DIM, V_DIM, D_DIM, 0, V_DIM, 1);

    const long SD = (long)S_LEN * D_DIM;

    // ---- embedding + positional, im2col, conv-as-GEMM ----
    embed_kernel<<<dim3((unsigned)((SD + 255) / 256)), dim3(256), 0, stream>>>(x, emb, pos, hbf);
    long x3Tot = SD * K_TAPS;
    im2col_kernel<<<dim3((unsigned)((x3Tot + 255) / 256)), dim3(256), 0, stream>>>(hbf, x3);
    gemm(x3, K_TAPS * D_DIM, wcv_p, convb, conv_out, D_DIM, S_LEN, D_DIM, K_TAPS * D_DIM);

    // ---- nodes = conv_out * sigmoid(conv_out @ wg1 + bg1) ----
    f32_to_bf16_kernel<<<dim3((unsigned)((SD + 255) / 256)), dim3(256), 0, stream>>>(conv_out, hbf, SD);
    gemm(hbf, D_DIM, wg1_p, bg1, gl_buf, D_DIM, S_LEN, D_DIM, D_DIM);
    nodes_kernel<<<dim3((unsigned)((SD + 255) / 256)), dim3(256), 0, stream>>>(conv_out, gl_buf, curA);

    // ---- binary-tree merge loop (11 sequential steps, double-buffered) ----
    float* cur = curA;
    float* nxt = curB;
    for (int step = 1; step < S_LEN; step <<= 1) {
        int  Sm      = S_LEN - step;
        long combTot = (long)Sm * 2 * D_DIM;
        comb_kernel<<<dim3((unsigned)((combTot + 255) / 256)), dim3(256), 0, stream>>>(cur, x3, Sm, step);

        gemm(x3,          2 * D_DIM, wmv_p, bmv, val_buf, D_DIM, Sm, D_DIM, 2 * D_DIM);
        gemm(x3,          2 * D_DIM, wmg_p, bmg, gl_buf,  D_DIM, Sm, D_DIM, 2 * D_DIM);
        gemm(x3,          2 * D_DIM, wrg_p, brg, rgl_buf, D_DIM, Sm, D_DIM, 2 * D_DIM);
        gemm(x3 + D_DIM,  2 * D_DIM, whw_p, bhw, hgl_buf, D_DIM, Sm, D_DIM, D_DIM);

        hipMemcpyAsync(nxt, cur, (size_t)step * D_DIM * sizeof(float),
                       hipMemcpyDeviceToDevice, stream);
        merge_kernel<<<dim3((unsigned)Sm), dim3(256), 0, stream>>>(
            cur, nxt, val_buf, gl_buf, rgl_buf, hgl_buf, rmsw, step);

        float* t = cur; cur = nxt; nxt = t;
    }

    // ---- final vocab projection ----
    f32_to_bf16_kernel<<<dim3((unsigned)((SD + 255) / 256)), dim3(256), 0, stream>>>(cur, hbf, SD);
    gemm(hbf, D_DIM, wpr_p, bpred, out, V_DIM, S_LEN, V_DIM, D_DIM);
}
